// TropicalConv1d_12335146074808
// MI455X (gfx1250) — compile-verified, matrix operand order checked by reading, not run on hardware
//
#include <hip/hip_runtime.h>
#include <cmath>

// ---------------------------------------------------------------------------
// Soft tropical conv1d:  y = log( conv1d(exp(x-xm), exp(w-wm)) ) + xm+wm + bias
// TEMP == 1.0 so divisions by T are omitted.
// Core conv: v_wmma_f32_16x16x32_f16 with LDS operands staged via
// GLOBAL_LOAD_ASYNC_TO_LDS_B128 (ASYNCcnt) -- CDNA5 / gfx1250, wave32.
// ---------------------------------------------------------------------------

#define USE_ASYNC 1

typedef _Float16 v8h  __attribute__((ext_vector_type(8)));
typedef _Float16 v16h __attribute__((ext_vector_type(16)));
typedef float    v8f  __attribute__((ext_vector_type(8)));

constexpr int B_   = 32;
constexpr int C_   = 128;   // CIN
constexpr int O_   = 128;   // COUT
constexpr int K_   = 7;
constexpr int L_   = 4096;
constexpr int PAD_ = 3;
constexpr int LP_  = L_ + 2 * PAD_;           // 4102 padded positions
constexpr unsigned NEG_INF_KEY = 0x007FFFFFu; // encode(-inf)

// Conv tiling
constexpr int MB_   = 256;                 // block M tile (positions)
constexpr int MROWS = MB_ + K_ - 1;        // 262 staged ex rows
constexpr int LDSP  = 136;                 // padded LDS row (halves): 272B
constexpr int EX_H  = MROWS * LDSP;        // 35632 halves
constexpr int EW_H  = O_ * LDSP;           // 17408 halves per k-tap buffer
constexpr size_t SMEM_BYTES = (size_t)(EX_H + 2 * EW_H) * 2;  // 140,896 B

// Workspace layout (bytes):
//   [0, 256)                  : 2x u32 max-reduction slots (ordered-key)
//   [256, 256+229376)         : ew_t  half [k][o][c]       (224 KB)
//   [229632, +33603584)       : ex_pad half [b][pos+6][c]  (~32 MB)
constexpr size_t WS_EWT_OFF = 256;
constexpr size_t WS_EX_OFF  = 229632;

// Monotone float <-> ordered-uint mapping (for atomicMax on floats)
__device__ __forceinline__ unsigned encodeF(float f) {
  unsigned b = __float_as_uint(f);
  return (b & 0x80000000u) ? ~b : (b | 0x80000000u);
}
__device__ __forceinline__ float decodeF(unsigned k) {
  unsigned b = (k & 0x80000000u) ? (k ^ 0x80000000u) : ~k;
  return __uint_as_float(b);
}

// ---- CDNA5 async global->LDS copy (16B per lane, tracked by ASYNCcnt) ------
__device__ __forceinline__ void async_ld_b128(_Float16* lds,
                                              const _Float16* gaddr) {
#if USE_ASYNC
  // Flat->LDS aperture keeps the LDS byte address in addr[31:0].
  unsigned loff = (unsigned)(unsigned long long)lds;
  asm volatile("global_load_async_to_lds_b128 %0, %1, off"
               :
               : "v"(loff), "v"(gaddr)
               : "memory");
#else
  *(v8h*)lds = *(const v8h*)gaddr;
#endif
}
__device__ __forceinline__ void wait_async() {
#if USE_ASYNC
  asm volatile("s_wait_asynccnt 0x0" ::: "memory");
#endif
}

// Build a 16-half fragment from two 16B LDS loads.
__device__ __forceinline__ v16h ldfrag(const _Float16* p) {
  v8h lo = *(const v8h*)(p);
  v8h hi = *(const v8h*)(p + 16);
  return __builtin_shufflevector(lo, hi, 0, 1, 2, 3, 4, 5, 6, 7, 8, 9, 10, 11,
                                 12, 13, 14, 15);
}

// ---------------------------------------------------------------------------
__global__ void trop_init_slots(unsigned* slots) {
  slots[0] = NEG_INF_KEY;
  slots[1] = NEG_INF_KEY;
}

__global__ void trop_reduce_max(const float* __restrict__ p, int n,
                                unsigned* __restrict__ slot) {
  __shared__ float sm[256];
  float m = -INFINITY;
  for (int i = blockIdx.x * blockDim.x + threadIdx.x; i < n;
       i += gridDim.x * blockDim.x)
    m = fmaxf(m, p[i]);
  sm[threadIdx.x] = m;
  __syncthreads();
  for (int s = 128; s > 0; s >>= 1) {
    if ((int)threadIdx.x < s)
      sm[threadIdx.x] = fmaxf(sm[threadIdx.x], sm[threadIdx.x + s]);
    __syncthreads();
  }
  if (threadIdx.x == 0) atomicMax(slot, encodeF(sm[0]));
}

// Zero the 3+3 halo rows of ex_pad (exp(-inf)=0 padding).
__global__ void trop_zero_pad(_Float16* __restrict__ ex) {
  int idx = blockIdx.x * blockDim.x + threadIdx.x;   // 32*6*128 = 24576
  int c = idx & 127;
  int r = (idx >> 7) % 6;
  int b = idx / (6 * 128);
  int pos = (r < 3) ? r : (LP_ - 6 + r);             // 0,1,2, 4099,4100,4101
  ex[((size_t)b * LP_ + pos) * C_ + c] = (_Float16)0.f;
}

// x[b][c][pos] (f32) -> ex_pad[b][pos+3][c] (f16), tiled LDS transpose + exp.
__global__ void trop_transform_x(const float* __restrict__ x,
                                 _Float16* __restrict__ ex,
                                 const unsigned* __restrict__ slots) {
  __shared__ float tile[32][33];
  const float xm = decodeF(slots[0]);
  const int b = blockIdx.z;
  const int posBase = blockIdx.x * 32;
  const int cBase = blockIdx.y * 32;
  const int tx = threadIdx.x, ty = threadIdx.y;      // block (32,8)
#pragma unroll
  for (int j = 0; j < 4; j++) {
    int c = cBase + ty + 8 * j;
    tile[ty + 8 * j][tx] = x[((size_t)b * C_ + c) * L_ + posBase + tx];
  }
  __syncthreads();
#pragma unroll
  for (int j = 0; j < 4; j++) {
    int pl = ty + 8 * j;
    ex[((size_t)b * LP_ + posBase + PAD_ + pl) * C_ + cBase + tx] =
        (_Float16)__expf(tile[tx][pl] - xm);
  }
}

// weight[o][c][k] (f32) -> ew_t[k][o][c] (f16)
__global__ void trop_transform_w(const float* __restrict__ w,
                                 _Float16* __restrict__ ewt,
                                 const unsigned* __restrict__ slots) {
  const float wm = decodeF(slots[1]);
  int idx = blockIdx.x * blockDim.x + threadIdx.x;   // 7*128*128 = 114688
  int c = idx & 127;
  int o = (idx >> 7) & 127;
  int k = idx >> 14;
  ewt[idx] = (_Float16)__expf(w[((size_t)o * C_ + c) * K_ + k] - wm);
}

// ---------------------------------------------------------------------------
// Main conv. Grid = B * (L/256) = 512 blocks, 256 threads (8 waves, 4Mx2N).
// Wave tile 64x64 -> 16 accumulators; per reduction step (k-tap, 32-ch chunk):
// 4 A-frags + 4 B-frags from LDS -> 16 v_wmma_f32_16x16x32_f16.
// LDS: ex tile (262x136 halves) + double-buffered one-tap weights (2x128x136),
// both filled with global_load_async_to_lds_b128 (prefetch overlaps WMMAs).
// ---------------------------------------------------------------------------
__global__ void __launch_bounds__(256)
trop_conv_wmma(const _Float16* __restrict__ ex,
               const _Float16* __restrict__ ewt,
               const float* __restrict__ bias,
               const unsigned* __restrict__ slots,
               float* __restrict__ out) {
  extern __shared__ _Float16 smem[];
  _Float16* ex_s = smem;                 // [MROWS][LDSP]
  _Float16* ew_s = smem + EX_H;          // [2][O_][LDSP]

  const float shift = decodeF(slots[0]) + decodeF(slots[1]);  // xm + wm
  const int b = blockIdx.x >> 4;         // 16 mBlocks per batch
  const int mBlock = blockIdx.x & 15;
  const int P0 = mBlock * MB_;           // first output position of block
  const int tid = threadIdx.x;
  const int wave = tid >> 5;
  const int lane = tid & 31;
  const int ln = lane & 15;
  const int hi = lane >> 4;              // K sub-stripe select
  const int mWave = (wave >> 1) * 64;    // wave M offset in block tile
  const int nWave = (wave & 1) * 64;     // wave N offset

  // ---- stage ex tile: 262 rows x 128 halves -> LDS rows of 136 (async) -----
  const _Float16* gx = ex + ((size_t)b * LP_ + P0) * C_;
  for (int i = tid; i < MROWS * 8; i += 256) {
    int row = i >> 3, ch = i & 7;
    async_ld_b128(&ex_s[row * LDSP + ch * 8], gx + row * C_ + ch * 8);
  }

  // ---- stage weights for k=0 into buffer 0 (async) -------------------------
  for (int i = tid; i < O_ * 8; i += 256) {
    int row = i >> 3, ch = i & 7;
    async_ld_b128(&ew_s[row * LDSP + ch * 8], ewt + row * C_ + ch * 8);
  }
  wait_async();
  __syncthreads();

  v8f acc[4][4] = {};

  for (int k = 0; k < K_; k++) {
    // prefetch next tap's weights into the other buffer (overlaps WMMAs)
    if (k < K_ - 1) {
      const _Float16* gw = ewt + (size_t)(k + 1) * O_ * C_;
      _Float16* dst = ew_s + ((k + 1) & 1) * EW_H;
      for (int i = tid; i < O_ * 8; i += 256) {
        int row = i >> 3, ch = i & 7;
        async_ld_b128(&dst[row * LDSP + ch * 8], gw + row * C_ + ch * 8);
      }
    }

    const _Float16* wbuf = ew_s + (k & 1) * EW_H;
    const _Float16* arow = &ex_s[(mWave + ln + k) * LDSP + hi * 8];
    const _Float16* brow = &wbuf[(nWave + ln) * LDSP + hi * 8];

#pragma unroll
    for (int cc = 0; cc < 4; cc++) {     // 4 chunks of 32 input channels
      v16h a[4], bf[4];
#pragma unroll
      for (int ms = 0; ms < 4; ms++)
        a[ms] = ldfrag(arow + ms * 16 * LDSP + cc * 32);
#pragma unroll
      for (int ns = 0; ns < 4; ns++)
        bf[ns] = ldfrag(brow + ns * 16 * LDSP + cc * 32);
#pragma unroll
      for (int ms = 0; ms < 4; ms++)
#pragma unroll
        for (int ns = 0; ns < 4; ns++)
          acc[ms][ns] = __builtin_amdgcn_wmma_f32_16x16x32_f16(
              false, a[ms], false, bf[ns], (short)0, acc[ms][ns], false, false);
    }
    wait_async();                        // prefetched tap landed in LDS
    __syncthreads();                     // publish; buffers safe to reuse
  }

  // ---- epilogue: lane holds N = nWave+ns*16+ln, rows M = (hi?8:0)+r --------
#pragma unroll
  for (int ns = 0; ns < 4; ns++) {
    const int o = nWave + ns * 16 + ln;
    const float bv = bias[o] + shift;
#pragma unroll
    for (int ms = 0; ms < 4; ms++) {
      float r[8];
#pragma unroll
      for (int i = 0; i < 8; i++) r[i] = __logf(acc[ms][ns][i]) + bv;
      const int pos0 = P0 + mWave + ms * 16 + hi * 8;
      float* op = out + ((size_t)(b * O_ + o)) * L_ + pos0;
      *(float4*)(op) = make_float4(r[0], r[1], r[2], r[3]);
      *(float4*)(op + 4) = make_float4(r[4], r[5], r[6], r[7]);
    }
  }
}

// ---------------------------------------------------------------------------
extern "C" void kernel_launch(void* const* d_in, const int* in_sizes, int n_in,
                              void* d_out, int out_size, void* d_ws,
                              size_t ws_size, hipStream_t stream) {
  const float* x = (const float*)d_in[0];      // (32,128,4096)
  const float* w = (const float*)d_in[1];      // (128,128,7)
  const float* bias = (const float*)d_in[2];   // (128,)
  float* out = (float*)d_out;                  // (32,128,4096) f32

  char* ws = (char*)d_ws;                      // needs ~32.3 MiB
  unsigned* slots = (unsigned*)ws;
  _Float16* ewt = (_Float16*)(ws + WS_EWT_OFF);
  _Float16* ex = (_Float16*)(ws + WS_EX_OFF);

  (void)in_sizes; (void)n_in; (void)out_size; (void)ws_size;

  trop_init_slots<<<1, 1, 0, stream>>>(slots);

  trop_reduce_max<<<1024, 256, 0, stream>>>(x, B_ * C_ * L_, slots + 0);
  trop_reduce_max<<<64, 256, 0, stream>>>(w, O_ * C_ * K_, slots + 1);

  trop_zero_pad<<<(B_ * 6 * C_) / 256, 256, 0, stream>>>(ex);
  trop_transform_x<<<dim3(L_ / 32, C_ / 32, B_), dim3(32, 8), 0, stream>>>(
      x, ex, slots);
  trop_transform_w<<<(K_ * O_ * C_) / 256, 256, 0, stream>>>(w, ewt, slots);

  trop_conv_wmma<<<B_ * (L_ / MB_), 256, SMEM_BYTES, stream>>>(ex, ewt, bias,
                                                               slots, out);
}
